// PointNetSetAbstractionMsg_82841329205663
// MI455X (gfx1250) — matrix operand order repaired and checked by reading, not compiled, not run on hardware
//
#include <hip/hip_runtime.h>
#include <hip/hip_bf16.h>

typedef __attribute__((ext_vector_type(16))) _Float16 v16h;
typedef __attribute__((ext_vector_type(8)))  float    v8f;
typedef __attribute__((ext_vector_type(4)))  unsigned u32x4;

#define NPTS   4096
#define NBATCH 8
#define NCENT  1024
#define OUT_CH 320

union F16x16u { v16h v; unsigned u[8]; u32x4 q[2]; _Float16 h[16]; };

// ---------------------------------------------------------------------------
// 1. Farthest point sampling: one block per batch, 256 threads, 16 pts/thread.
// ---------------------------------------------------------------------------
__global__ __launch_bounds__(256) void fps_kernel(const float* __restrict__ xyz,
                                                  int* __restrict__ fpsIdx) {
  const int b = blockIdx.x;
  const int tid = threadIdx.x;
  const float* X = xyz + (size_t)b * 3 * NPTS;
  float px[16], py[16], pz[16], dist[16];
#pragma unroll
  for (int i = 0; i < 16; ++i) {
    int n = i * 256 + tid;
    px[i] = X[n]; py[i] = X[NPTS + n]; pz[i] = X[2 * NPTS + n];
    dist[i] = 1e10f;
  }
  __shared__ float redV[256];
  __shared__ int   redI[256];
  __shared__ int   farS;
  if (tid == 0) farS = 0;
  __syncthreads();
  for (int it = 0; it < NCENT; ++it) {
    int far = farS;
    if (tid == 0) fpsIdx[b * NCENT + it] = far;
    float cx = X[far], cy = X[NPTS + far], cz = X[2 * NPTS + far];
    float best = -1.f; int bi = 0;
#pragma unroll
    for (int i = 0; i < 16; ++i) {
      float dx = px[i] - cx, dy = py[i] - cy, dz = pz[i] - cz;
      float d = dx * dx + dy * dy + dz * dz;
      float nd = fminf(dist[i], d);
      dist[i] = nd;
      int n = i * 256 + tid;
      if (nd > best) { best = nd; bi = n; }
    }
    redV[tid] = best; redI[tid] = bi;
    __syncthreads();
    for (int s = 128; s > 0; s >>= 1) {
      if (tid < s) {
        float v2 = redV[tid + s]; int i2 = redI[tid + s];
        if (v2 > redV[tid] || (v2 == redV[tid] && i2 < redI[tid])) {
          redV[tid] = v2; redI[tid] = i2;
        }
      }
      __syncthreads();
    }
    if (tid == 0) farS = redI[0];
    __syncthreads();
  }
}

// ---------------------------------------------------------------------------
// 2. Gather new_xyz (B,S,3) into ws and transposed (B,3,S) into d_out.
// ---------------------------------------------------------------------------
__global__ void gather_newxyz(const float* __restrict__ xyz, const int* __restrict__ fpsIdx,
                              float* __restrict__ nxyz, float* __restrict__ outXyz) {
  int t = blockIdx.x * blockDim.x + threadIdx.x;
  if (t >= NBATCH * NCENT) return;
  int b = t >> 10, s = t & 1023;
  int j = fpsIdx[t];
#pragma unroll
  for (int c = 0; c < 3; ++c) {
    float v = xyz[((size_t)b * 3 + c) * NPTS + j];
    nxyz[(size_t)t * 3 + c] = v;
    outXyz[((size_t)b * 3 + c) * NCENT + s] = v;
  }
}

// ---------------------------------------------------------------------------
// 3. Ball query: one wave32 per centroid; ballot compaction in index order.
// ---------------------------------------------------------------------------
__global__ __launch_bounds__(256) void ball_query(const float* __restrict__ xyz,
                                                  const float* __restrict__ nxyz,
                                                  int* __restrict__ idxOut,
                                                  float r2, int ns) {
  int gw = (blockIdx.x * blockDim.x + threadIdx.x) >> 5;
  int lane = threadIdx.x & 31;
  if (gw >= NBATCH * NCENT) return;
  int b = gw >> 10;
  const float* X = xyz + (size_t)b * 3 * NPTS;
  float cx = nxyz[(size_t)gw * 3 + 0], cy = nxyz[(size_t)gw * 3 + 1], cz = nxyz[(size_t)gw * 3 + 2];
  int* out = idxOut + (size_t)gw * ns;
  int cnt = 0;
  int first = NPTS - 1;   // JAX clamps OOB index N -> N-1
  bool haveFirst = false;
  for (int base = 0; base < NPTS && cnt < ns; base += 32) {
    int n = base + lane;
    float dx = X[n] - cx, dy = X[NPTS + n] - cy, dz = X[2 * NPTS + n] - cz;
    float d = fmaf(dx, dx, fmaf(dy, dy, dz * dz));
    bool in = (d <= r2);
    unsigned mask = (unsigned)__ballot(in);
    if (!haveFirst && mask) { first = base + __ffs(mask) - 1; haveFirst = true; }
    int pre = __popc(mask & ((1u << lane) - 1u));
    if (in && cnt + pre < ns) out[cnt + pre] = n;
    cnt += __popc(mask);
  }
  if (cnt > ns) cnt = ns;
  for (int k = cnt + lane; k < ns; k += 32) out[k] = first;
}

// ---------------------------------------------------------------------------
// 4. Gather grouped features into f16 rows of 32 (cols 0-2 rel-xyz, 3-8 feats,
//    9-31 zero pad so layer-0 uses a single K=32 WMMA tile).
// ---------------------------------------------------------------------------
__global__ void gather_group(const float* __restrict__ xyz, const float* __restrict__ pts,
                             const float* __restrict__ nxyz, const int* __restrict__ idx,
                             _Float16* __restrict__ G, int nsShift, int Mtotal) {
  int m = blockIdx.x * blockDim.x + threadIdx.x;
  if (m >= Mtotal) return;
  int g = m >> nsShift;
  int b = g >> 10;
  int j = idx[m];
  const float* X = xyz + (size_t)b * 3 * NPTS;
  F16x16u lo, hi;
#pragma unroll
  for (int k = 0; k < 16; ++k) { lo.h[k] = (_Float16)0.f; hi.h[k] = (_Float16)0.f; }
  lo.h[0] = (_Float16)(X[j]            - nxyz[(size_t)g * 3 + 0]);
  lo.h[1] = (_Float16)(X[NPTS + j]     - nxyz[(size_t)g * 3 + 1]);
  lo.h[2] = (_Float16)(X[2 * NPTS + j] - nxyz[(size_t)g * 3 + 2]);
#pragma unroll
  for (int d = 0; d < 6; ++d)
    lo.h[3 + d] = (_Float16)pts[((size_t)b * 6 + d) * NPTS + j];
  _Float16* dst = G + (size_t)m * 32;
  *(v16h*)dst        = lo.v;
  *(v16h*)(dst + 16) = hi.v;
}

// ---------------------------------------------------------------------------
// 5. Weight prep: fp32 (Cin,Cout) row-major -> f16 B-fragment layout.
//    Fragment (kt,nt), lane l, elem i -> K = kt*32 + (l>>4)*8 + (i<8 ? i : i+8), N = nt*16 + (l&15)
// ---------------------------------------------------------------------------
__global__ void prep_weights(const float* __restrict__ W, int Cin, int K, int Cout,
                             _Float16* __restrict__ wf) {
  int e = blockIdx.x * blockDim.x + threadIdx.x;
  int NT = Cout >> 4;
  int total = (K >> 5) * NT * 512;
  if (e >= total) return;
  int frag = e >> 9, within = e & 511;
  int lane = within >> 4, i = within & 15;
  int kt = frag / NT, nt = frag - kt * NT;
  int kk = kt * 32 + ((lane >> 4) << 3) + (i < 8 ? i : i + 8);
  int nn = nt * 16 + (lane & 15);
  float v = (kk < Cin) ? W[(size_t)kk * Cout + nn] : 0.f;
  wf[e] = (_Float16)v;
}

// ---------------------------------------------------------------------------
// 6. BN stats finalize: scale = g*rsqrt(var+eps), shift = beta - mean*scale.
// ---------------------------------------------------------------------------
__global__ void finalize_stats(const float* __restrict__ stats, const float* __restrict__ gamma,
                               const float* __restrict__ beta, float invM, int C,
                               float* __restrict__ scale, float* __restrict__ shift) {
  int c = blockIdx.x * blockDim.x + threadIdx.x;
  if (c >= C) return;
  float mean = stats[c] * invM;
  float var  = stats[C + c] * invM - mean * mean;
  float sc = gamma[c] * rsqrtf(var + 1e-5f);
  scale[c] = sc;
  shift[c] = beta[c] - mean * sc;
}

// ---------------------------------------------------------------------------
// 7. WMMA branch pass. Each wave owns a 16-row M-tile. Layer-0 A-fragments are
//    loaded straight from global memory (G rows are fragment-friendly: two
//    16-byte chunks per lane), later layers hand off through a private LDS
//    tile using the documented CDNA5 f16 A-fragment layout.
//    MODE: 0 = BN+ReLU -> LDS, 1 = stats reduce, 2 = BN+ReLU -> max -> d_out.
// ---------------------------------------------------------------------------
struct PassArgs {
  const _Float16* G;
  const _Float16* wf0; const _Float16* wf1; const _Float16* wf2;
  const float* bias0; const float* bias1; const float* bias2;
  const float* scale0; const float* shift0;
  const float* scale1; const float* shift1;
  const float* scale2; const float* shift2;
  float* stats;
  float* outp;          // new_points base in d_out
  int Mtiles;
  int outChanBase;
};

template<int K, int NOUT, int STRH, int MODE, bool DIRECT>
__device__ __forceinline__ void layer_step(_Float16* tile, int lane, v16h aDirect,
                                           const _Float16* __restrict__ wf,
                                           const float* __restrict__ bias,
                                           const float* __restrict__ scale,
                                           const float* __restrict__ shift,
                                           float* ssum, float* ssq,
                                           float* __restrict__ outp,
                                           int outChanBase, int g) {
  constexpr int KT = K / 32, NT = NOUT / 16;
  const int row = lane & 15;
  const int hi  = lane >> 4;
  // A fragments (16x32 f16 layout): lanes 0-15 rows, K = (i<8?i:i+8) + 8*hi
  v16h a[KT];
  if constexpr (DIRECT) {
    a[0] = aDirect;                       // K == 32, fragment preloaded from global
  } else {
#pragma unroll
    for (int kt = 0; kt < KT; ++kt) {
      F16x16u r;
      const _Float16* p = tile + row * STRH + kt * 32 + hi * 8;
#pragma unroll
      for (int v = 0; v < 8; ++v) {
        int k2 = 2 * v; if (k2 >= 8) k2 += 8;
        r.u[v] = *(const unsigned*)(p + k2);
      }
      a[kt] = r.v;
    }
  }
#pragma unroll
  for (int nt = 0; nt < NT; ++nt) {
    v8f acc = {0.f, 0.f, 0.f, 0.f, 0.f, 0.f, 0.f, 0.f};
#pragma unroll
    for (int kt = 0; kt < KT; ++kt) {
      v16h bw = *(const v16h*)(wf + ((size_t)(kt * NT + nt) * 32 + lane) * 16);
      acc = __builtin_amdgcn_wmma_f32_16x16x32_f16(false, a[kt], false, bw,
                                                   (short)0, acc, false, false);
    }
    const int c = nt * 16 + row;           // one channel per lane (C/D layout)
    const float bb = bias[c];
    if constexpr (MODE == 0) {
      const float sc = scale[c], sh = shift[c];
#pragma unroll
      for (int r8 = 0; r8 < 8; ++r8) {
        float y = sc * (acc[r8] + bb) + sh;
        y = y > 0.f ? y : 0.f;
        tile[(r8 + 8 * hi) * STRH + c] = (_Float16)y;
      }
    } else if constexpr (MODE == 1) {
      float s = 0.f, q = 0.f;
#pragma unroll
      for (int r8 = 0; r8 < 8; ++r8) { float y = acc[r8] + bb; s += y; q += y * y; }
      ssum[nt] += s; ssq[nt] += q;
    } else {
      const float sc = scale[c], sh = shift[c];
      float m = 0.f;   // ReLU >= 0, so 0 is the identity for the max
#pragma unroll
      for (int r8 = 0; r8 < 8; ++r8) m = fmaxf(m, sc * (acc[r8] + bb) + sh);
      m = fmaxf(m, __shfl_xor(m, 16, 32));
      if (hi == 0) {
        size_t o = ((size_t)((g >> 10) * OUT_CH + outChanBase + c)) * NCENT + (size_t)(g & 1023);
        atomicMax((unsigned*)(outp + o), __float_as_uint(m));  // monotone for floats >= 0
      }
    }
  }
}

template<int C1, int C2, int C3, int NS, int LT, bool STATS>
__global__ __launch_bounds__(128) void branch_pass(PassArgs pa) {
  constexpr int MAXC  = (C3 > 32 ? C3 : 32);
  constexpr int STRH  = MAXC + 4;                       // even -> 4B aligned u32 LDS ops
  constexpr int CL    = (LT == 0 ? C1 : (LT == 1 ? C2 : C3));
  constexpr int NTL   = CL / 16;
  constexpr int NSH   = (NS == 16) ? 4 : ((NS == 32) ? 5 : 7);
  __shared__ _Float16 tiles[4][16 * STRH];
  const int wid  = threadIdx.x >> 5;
  const int lane = threadIdx.x & 31;
  _Float16* tile = tiles[wid];
  const int row = lane & 15, hi = lane >> 4;

  float ssum[NTL] = {};
  float ssq [NTL] = {};

  const int totalWaves = gridDim.x * 4;
  for (int t = blockIdx.x * 4 + wid; t < pa.Mtiles; t += totalWaves) {
    // Layer-0 A fragment straight from global: G rows (32 halves, 64B) hold the
    // required halves as two contiguous 16B chunks per lane:
    //   i<8  -> K = i + 8*hi      == gp[8*hi + i]
    //   i>=8 -> K = i + 8 + 8*hi  == gp[16 + 8*hi + (i-8)]
    v16h a0;
    {
      const _Float16* gp = pa.G + ((size_t)t * 16 + row) * 32;
      F16x16u r;
      r.q[0] = *(const u32x4*)(gp + hi * 8);
      r.q[1] = *(const u32x4*)(gp + 16 + hi * 8);
      a0 = r.v;
    }
    const int g = (t * 16) >> NSH;   // one (b,s) group per tile (16 | NS)

    layer_step<32, C1, STRH, (LT == 0 ? (STATS ? 1 : 2) : 0), true>(
        tile, lane, a0, pa.wf0, pa.bias0, pa.scale0, pa.shift0, ssum, ssq, pa.outp, pa.outChanBase, g);
    if constexpr (LT >= 1)
      layer_step<C1, C2, STRH, (LT == 1 ? (STATS ? 1 : 2) : 0), false>(
          tile, lane, a0, pa.wf1, pa.bias1, pa.scale1, pa.shift1, ssum, ssq, pa.outp, pa.outChanBase, g);
    if constexpr (LT >= 2)
      layer_step<C2, C3, STRH, (LT == 2 ? (STATS ? 1 : 2) : 0), false>(
          tile, lane, a0, pa.wf2, pa.bias2, pa.scale2, pa.shift2, ssum, ssq, pa.outp, pa.outChanBase, g);
  }

  if constexpr (STATS) {
#pragma unroll
    for (int nt = 0; nt < NTL; ++nt) {
      float s = ssum[nt] + __shfl_xor(ssum[nt], 16, 32);
      float q = ssq[nt]  + __shfl_xor(ssq[nt],  16, 32);
      if (hi == 0) {
        atomicAdd(&pa.stats[nt * 16 + row], s);
        atomicAdd(&pa.stats[CL + nt * 16 + row], q);
      }
    }
  }
}

// ---------------------------------------------------------------------------
// Host-side pass sequencing for one branch.
// ---------------------------------------------------------------------------
template<int C1, int C2, int C3, int NS>
static void run_branch(PassArgs pa, float* const* stats, float* const* scale, float* const* shift,
                       const float* const* gamma, const float* const* beta,
                       float invM, int blocks, hipStream_t stream) {
  pa.stats = stats[0];
  hipMemsetAsync(stats[0], 0, 2 * C1 * sizeof(float), stream);
  branch_pass<C1, C2, C3, NS, 0, true><<<blocks, 128, 0, stream>>>(pa);
  finalize_stats<<<1, C1, 0, stream>>>(stats[0], gamma[0], beta[0], invM, C1, scale[0], shift[0]);

  pa.stats = stats[1];
  hipMemsetAsync(stats[1], 0, 2 * C2 * sizeof(float), stream);
  branch_pass<C1, C2, C3, NS, 1, true><<<blocks, 128, 0, stream>>>(pa);
  finalize_stats<<<1, C2, 0, stream>>>(stats[1], gamma[1], beta[1], invM, C2, scale[1], shift[1]);

  pa.stats = stats[2];
  hipMemsetAsync(stats[2], 0, 2 * C3 * sizeof(float), stream);
  branch_pass<C1, C2, C3, NS, 2, true><<<blocks, 128, 0, stream>>>(pa);
  finalize_stats<<<1, C3, 0, stream>>>(stats[2], gamma[2], beta[2], invM, C3, scale[2], shift[2]);

  branch_pass<C1, C2, C3, NS, 2, false><<<blocks, 128, 0, stream>>>(pa);
}

extern "C" void kernel_launch(void* const* d_in, const int* in_sizes, int n_in,
                              void* d_out, int out_size, void* d_ws, size_t ws_size,
                              hipStream_t stream) {
  (void)in_sizes; (void)n_in; (void)ws_size;
  const float* xyz = (const float*)d_in[0];   // (B,3,N)
  const float* pts = (const float*)d_in[1];   // (B,6,N)

  char* ws = (char*)d_ws;
  size_t off = 0;
  auto take = [&](size_t bytes) -> void* {
    size_t cur = (off + 255) & ~(size_t)255;
    off = cur + bytes;
    return (void*)(ws + cur);
  };
  int*      fpsIdx = (int*)     take((size_t)NBATCH * NCENT * 4);
  float*    nxyz   = (float*)   take((size_t)NBATCH * NCENT * 3 * 4);
  int*      idxBuf = (int*)     take((size_t)NBATCH * NCENT * 128 * 4);       // reused per branch
  _Float16* G      = (_Float16*)take((size_t)NBATCH * NCENT * 128 * 32 * 2);  // reused per branch
  _Float16* wfAll  = (_Float16*)take(65536 * 2);
  float*    statsB = (float*)   take(3 * 256 * 4);
  float*    scsh   = (float*)   take(3 * 256 * 4);

  float* outNewXyz = (float*)d_out;
  float* outNewPts = (float*)d_out + NBATCH * 3 * NCENT;

  hipMemsetAsync(d_out, 0, (size_t)out_size * sizeof(float), stream);

  fps_kernel<<<NBATCH, 256, 0, stream>>>(xyz, fpsIdx);
  gather_newxyz<<<(NBATCH * NCENT + 255) / 256, 256, 0, stream>>>(xyz, fpsIdx, nxyz, outNewXyz);

  const int   C1s[3] = {32, 64, 64}, C2s[3] = {32, 64, 96}, C3s[3] = {64, 128, 128};
  const int   NSs[3] = {16, 32, 128}, NSsh[3] = {4, 5, 7};
  const float R2s[3] = {0.01f, 0.04f, 0.16f};
  const int   outBase[3] = {0, 64, 192};

  for (int br = 0; br < 3; ++br) {
    const int ns = NSs[br];
    const int M = NBATCH * NCENT * ns;
    const int Mtiles = M / 16;

    ball_query<<<(NBATCH * NCENT * 32) / 256, 256, 0, stream>>>(xyz, nxyz, idxBuf, R2s[br], ns);
    gather_group<<<(M + 255) / 256, 256, 0, stream>>>(xyz, pts, nxyz, idxBuf, G, NSsh[br], M);

    const float* W[3]; const float* Bs[3]; const float* Gm[3]; const float* Bt[3];
    for (int ly = 0; ly < 3; ++ly) {
      int base = 2 + (br * 3 + ly) * 4;
      W[ly]  = (const float*)d_in[base + 0];
      Bs[ly] = (const float*)d_in[base + 1];
      Gm[ly] = (const float*)d_in[base + 2];
      Bt[ly] = (const float*)d_in[base + 3];
    }
    const int Cin[4] = {9, C1s[br], C2s[br], C3s[br]};
    const int Kp[3]  = {32, C1s[br], C2s[br]};
    _Float16* wf[3];
    size_t woff = 0;
    for (int ly = 0; ly < 3; ++ly) {
      wf[ly] = wfAll + woff;
      int total = (Kp[ly] / 32) * (Cin[ly + 1] / 16) * 512;
      woff += (size_t)total;
      prep_weights<<<(total + 255) / 256, 256, 0, stream>>>(W[ly], Cin[ly], Kp[ly], Cin[ly + 1], wf[ly]);
    }

    float* stats[3]; float* scale[3]; float* shift[3];
    for (int ly = 0; ly < 3; ++ly) {
      stats[ly] = statsB + ly * 256;
      scale[ly] = scsh + ly * 256;
      shift[ly] = scsh + ly * 256 + 128;
    }

    PassArgs pa{};
    pa.G = G;
    pa.wf0 = wf[0]; pa.wf1 = wf[1]; pa.wf2 = wf[2];
    pa.bias0 = Bs[0]; pa.bias1 = Bs[1]; pa.bias2 = Bs[2];
    pa.scale0 = scale[0]; pa.shift0 = shift[0];
    pa.scale1 = scale[1]; pa.shift1 = shift[1];
    pa.scale2 = scale[2]; pa.shift2 = shift[2];
    pa.outp = outNewPts;
    pa.Mtiles = Mtiles;
    pa.outChanBase = outBase[br];

    float invM = 1.f / (float)M;
    int blocks = (Mtiles + 3) / 4;
    if (blocks > 4096) blocks = 4096;

    if (br == 0)      run_branch<32, 32, 64, 16 >(pa, stats, scale, shift, Gm, Bt, invM, blocks, stream);
    else if (br == 1) run_branch<64, 64, 128, 32>(pa, stats, scale, shift, Gm, Bt, invM, blocks, stream);
    else              run_branch<64, 96, 128, 128>(pa, stats, scale, shift, Gm, Bt, invM, blocks, stream);
  }
}
